// Net_12781822673244
// MI455X (gfx1250) — compile-verified, matrix operand order checked by reading, not run on hardware
//
#include <hip/hip_runtime.h>
#include <math.h>

typedef __attribute__((ext_vector_type(16))) _Float16 v16h;
typedef __attribute__((ext_vector_type(8)))  float    v8f;

#define F_IN 512
#define HID  16

// ---------------- elementwise helpers ----------------

__global__ void fill_kernel(float* __restrict__ p, float v, int n) {
    int i = blockIdx.x * blockDim.x + threadIdx.x;
    if (i < n) p[i] = v;
}

__global__ void deg_kernel(const int* __restrict__ dst, float* __restrict__ deg, int E) {
    int i = blockIdx.x * blockDim.x + threadIdx.x;
    if (i < E) atomicAdd(&deg[dst[i]], 1.0f);
}

__global__ void rsqrt_kernel(const float* __restrict__ deg, float* __restrict__ dinv, int n) {
    int i = blockIdx.x * blockDim.x + threadIdx.x;
    if (i < n) dinv[i] = rsqrtf(deg[i]);
}

// ---------------- WMMA GEMM, layer 1: [N,512] x [512,16] ----------------
// One wave computes one 16x16 output tile. A fragment layout (16-bit A, 16x32):
//   lane l: M = l&15, kbase = (l>>4)*8; VGPR v holds K = (v<4?0:16)+kbase+(v&3)*2 (pair).
// B fragment mirrors with N = l&15. C/D: VGPR v, lanes<16 -> M=v, lanes>=16 -> M=v+8.
__global__ __launch_bounds__(128) void gemm1_kernel(const float* __restrict__ A,
                                                    const float* __restrict__ W,
                                                    float* __restrict__ out, int nTiles) {
    int wave = threadIdx.x >> 5;
    int tile = blockIdx.x * 4 + wave;
    if (tile >= nTiles) return;            // wave-uniform: EXEC stays all-1s for WMMA
    int lane = threadIdx.x & 31;
    int mn   = lane & 15;                  // M for A, N for B
    int kb   = (lane >> 4) << 3;
    int rowBase = tile << 4;
    const float* ap = A + (size_t)(rowBase + mn) * F_IN;
    v8f acc = {};
    for (int k0 = 0; k0 < F_IN; k0 += 32) {
        v16h a, b;
        #pragma unroll
        for (int v = 0; v < 8; ++v) {
            int K = ((v >> 2) << 4) + kb + ((v & 3) << 1);
            float2 f = *(const float2*)(ap + k0 + K);
            a[2 * v]     = (_Float16)f.x;
            a[2 * v + 1] = (_Float16)f.y;
            b[2 * v]     = (_Float16)W[(size_t)(k0 + K) * HID + mn];
            b[2 * v + 1] = (_Float16)W[(size_t)(k0 + K + 1) * HID + mn];
        }
        acc = __builtin_amdgcn_wmma_f32_16x16x32_f16(false, a, false, b,
                                                     (short)0, acc, false, false);
    }
    int half = lane >> 4;
    #pragma unroll
    for (int v = 0; v < 8; ++v) {
        int row = rowBase + v + half * 8;
        out[(size_t)row * HID + mn] = acc[v];
    }
}

// ---------------- WMMA GEMM, layer 2: [N,16] x [16,16], K zero-padded to 32 ----------------
__global__ __launch_bounds__(128) void gemm2_kernel(const float* __restrict__ A,
                                                    const float* __restrict__ W,
                                                    float* __restrict__ out, int nTiles) {
    int wave = threadIdx.x >> 5;
    int tile = blockIdx.x * 4 + wave;
    if (tile >= nTiles) return;
    int lane = threadIdx.x & 31;
    int mn   = lane & 15;
    int kb   = (lane >> 4) << 3;
    int rowBase = tile << 4;
    const float* ap = A + (size_t)(rowBase + mn) * HID;
    v16h a = {}, b = {};
    #pragma unroll
    for (int v = 0; v < 8; ++v) {
        int K = ((v >> 2) << 4) + kb + ((v & 3) << 1);
        if (K < HID) {   // K is even; K<16 implies K+1<16. VGPRs 4..7 stay zero (padding).
            float2 f = *(const float2*)(ap + K);
            a[2 * v]     = (_Float16)f.x;
            a[2 * v + 1] = (_Float16)f.y;
            b[2 * v]     = (_Float16)W[(size_t)K * HID + mn];
            b[2 * v + 1] = (_Float16)W[(size_t)(K + 1) * HID + mn];
        }
    }
    v8f acc = {};
    acc = __builtin_amdgcn_wmma_f32_16x16x32_f16(false, a, false, b,
                                                 (short)0, acc, false, false);
    int half = lane >> 4;
    #pragma unroll
    for (int v = 0; v < 8; ++v) {
        int row = rowBase + v + half * 8;
        out[(size_t)row * HID + mn] = acc[v];
    }
}

// ---------------- edge scatter: agg[dst] += feat[src] * dinv[src]*dinv[dst] ----------------
// 4 threads per edge; each gathers a float4 of the source row and does 4 f32 atomics.
// Tables are L2-resident (6.4MB each), so atomics resolve at L2.
__global__ void agg_kernel(const int* __restrict__ src, const int* __restrict__ dst,
                           const float* __restrict__ dinv, const float* __restrict__ feat,
                           float* __restrict__ agg, int E) {
    int tid = blockIdx.x * blockDim.x + threadIdx.x;
    if (tid >= E * 4) return;
    int e = tid >> 2, part = tid & 3;
    int s = src[e], d = dst[e];
    float norm = dinv[s] * dinv[d];
    float4 f = ((const float4*)(feat + (size_t)s * HID))[part];
    float* o = agg + (size_t)d * HID + part * 4;
    atomicAdd(o + 0, f.x * norm);
    atomicAdd(o + 1, f.y * norm);
    atomicAdd(o + 2, f.z * norm);
    atomicAdd(o + 3, f.w * norm);
}

// ---------------- self-loop + bias + ReLU ----------------
__global__ void bias_relu_kernel(const float* __restrict__ agg, const float* __restrict__ h,
                                 const float* __restrict__ dinv, const float* __restrict__ b,
                                 float* __restrict__ out, int total) {
    int tid = blockIdx.x * blockDim.x + threadIdx.x;
    if (tid >= total) return;
    int i = tid >> 4, c = tid & 15;
    float di = dinv[i];
    float v = agg[tid] + h[tid] * di * di + b[c];
    out[tid] = fmaxf(v, 0.0f);
}

// ---------------- self-loop + bias + log_softmax (16 classes) ----------------
__global__ void logsoftmax_kernel(const float* __restrict__ agg, const float* __restrict__ g,
                                  const float* __restrict__ dinv, const float* __restrict__ b,
                                  float* __restrict__ out, int n) {
    int i = blockIdx.x * blockDim.x + threadIdx.x;
    if (i >= n) return;
    float di = dinv[i];
    float di2 = di * di;
    const float* ga = agg + (size_t)i * HID;
    const float* gg = g + (size_t)i * HID;
    float v[HID];
    float m = -INFINITY;
    #pragma unroll
    for (int c = 0; c < HID; ++c) {
        v[c] = ga[c] + gg[c] * di2 + b[c];
        m = fmaxf(m, v[c]);
    }
    float s = 0.0f;
    #pragma unroll
    for (int c = 0; c < HID; ++c) s += __expf(v[c] - m);
    float lse = m + __logf(s);
    float* o = out + (size_t)i * HID;
    #pragma unroll
    for (int c = 0; c < HID; ++c) o[c] = v[c] - lse;
}

extern "C" void kernel_launch(void* const* d_in, const int* in_sizes, int n_in,
                              void* d_out, int out_size, void* d_ws, size_t ws_size,
                              hipStream_t stream) {
    const float* x  = (const float*)d_in[0];
    const float* W1 = (const float*)d_in[1];
    const float* b1 = (const float*)d_in[2];
    const float* W2 = (const float*)d_in[3];
    const float* b2 = (const float*)d_in[4];
    const int*   ei = (const int*)d_in[5];

    int N = in_sizes[0] / F_IN;   // 100000
    int E = in_sizes[5] / 2;      // 3200000
    const int* src = ei;
    const int* dst = ei + E;

    float* ws   = (float*)d_ws;
    float* deg  = ws;                       // [N]
    float* dinv = ws + (size_t)N;           // [N]
    float* h1   = ws + (size_t)2  * N;      // [N,16]
    float* agg1 = ws + (size_t)18 * N;      // [N,16]
    float* h2   = ws + (size_t)34 * N;      // [N,16]
    float* g    = h1;                       // reuse: h1 dead after bias_relu
    float* agg2 = agg1;                     // reuse: agg1 dead after bias_relu (re-zeroed)
    float* out  = (float*)d_out;

    const int B = 256;
    int nTiles = (N + 15) / 16;

    fill_kernel<<<(N + B - 1) / B, B, 0, stream>>>(deg, 1.0f, N);          // self-loop deg
    fill_kernel<<<(16 * N + B - 1) / B, B, 0, stream>>>(agg1, 0.0f, 16 * N);
    deg_kernel<<<(E + B - 1) / B, B, 0, stream>>>(dst, deg, E);
    rsqrt_kernel<<<(N + B - 1) / B, B, 0, stream>>>(deg, dinv, N);

    gemm1_kernel<<<(nTiles + 3) / 4, 128, 0, stream>>>(x, W1, h1, nTiles);
    agg_kernel<<<(4 * E + B - 1) / B, B, 0, stream>>>(src, dst, dinv, h1, agg1, E);
    bias_relu_kernel<<<(16 * N + B - 1) / B, B, 0, stream>>>(agg1, h1, dinv, b1, h2, 16 * N);

    gemm2_kernel<<<(nTiles + 3) / 4, 128, 0, stream>>>(h2, W2, g, nTiles);
    fill_kernel<<<(16 * N + B - 1) / B, B, 0, stream>>>(agg2, 0.0f, 16 * N);
    agg_kernel<<<(4 * E + B - 1) / B, B, 0, stream>>>(src, dst, dinv, g, agg2, E);
    logsoftmax_kernel<<<(N + B - 1) / B, B, 0, stream>>>(agg2, g, dinv, b2, out, N);
}
